// ScaledDotProductAttention_19344532702173
// MI455X (gfx1250) — matrix operand order repaired and checked by reading, not compiled, not run on hardware
//
#include <hip/hip_runtime.h>
#include <math.h>

typedef __attribute__((ext_vector_type(16))) _Float16 v16h;
typedef __attribute__((ext_vector_type(8)))  _Float16 v8h;
typedef __attribute__((ext_vector_type(4)))  _Float16 v4h;
typedef __attribute__((ext_vector_type(8)))  float    v8f;

#define WMMA_F16(a, b, c) \
  __builtin_amdgcn_wmma_f32_16x16x32_f16(false, (a), false, (b), (short)0, (c), false, false)

constexpr int Bb   = 8;     // batch
constexpr int Nn   = 4096;  // sequence length
constexpr int Dd   = 64;    // head dim
constexpr int QT   = 128;   // queries per block (8 waves x 16)
constexpr int NTHR = 256;   // 8 wave32
constexpr int KT   = 32;    // keys per iteration
constexpr int KSTR = 72;    // LDS row stride (halves) for K tile  (16B-multiple, bank-staggered)
constexpr int VSTR = 40;    // LDS row stride (halves) for V^T tile (16B-multiple)

// Native exp2: v_exp_f32 IS exp2 on AMD; skip libm's denorm guard (args are <= 0,
// flush-to-zero for very negative args is exactly what softmax wants).
__device__ __forceinline__ float fast_exp2(float x) {
#if __has_builtin(__builtin_amdgcn_exp2f)
  return __builtin_amdgcn_exp2f(x);
#else
  return __expf(x * 0.69314718055994530942f);
#endif
}

// Build a 16-bit A-matrix (16 x K-slice-of-32) fragment from an LDS row.
// Contiguous K layout: lane(h=lane>>4, m=lane&15): a[i] = A[m][16*h + i]
__device__ __forceinline__ v16h load_a16(const _Float16* rowp, int h) {
  v8h lo = *(const v8h*)(rowp + 16 * h);       // K = 16h .. 16h+7
  v8h hi = *(const v8h*)(rowp + 16 * h + 8);   // K = 16h+8 .. 16h+15
  v16h a;
#pragma unroll
  for (int i = 0; i < 8; ++i) { a[i] = lo[i]; a[i + 8] = hi[i]; }
  return a;
}

__global__ __launch_bounds__(NTHR, 1)
void fa_sT_kernel(const float* __restrict__ q,
                  const float* __restrict__ k,
                  const float* __restrict__ v,
                  float* __restrict__ out) {
  __shared__ alignas(16) _Float16 Kh[2][KT][KSTR];   // K tile, f16 row-major [key][d]
  __shared__ alignas(16) _Float16 Vt[2][Dd][VSTR];   // V tile, f16 TRANSPOSED [d][key]

  const int tid  = threadIdx.x;
  const int lane = tid & 31;
  const int wave = tid >> 5;             // 0..7
  const int r15  = lane & 15;            // row/col index within 16-group
  const int g    = lane >> 4;            // which 16-lane half
  const int bb   = blockIdx.y;
  const int q0   = blockIdx.x * QT + wave * 16;

  // ---- Q fragments (B-operand for S^T = K * Q^T), loop-invariant ----
  // Fold log2(e) into Q so S^T comes out in log2 units -> raw exp2 in softmax.
  const float LOG2E = 1.44269504088896340736f;
  const float* Qrow = q + ((size_t)bb * Nn + q0 + r15) * Dd;
  v16h qb[2];
#pragma unroll
  for (int s = 0; s < 2; ++s) {
    const float* src = Qrow + s * 32 + 16 * g;
#pragma unroll
    for (int c = 0; c < 4; ++c) {
      float4 f = *(const float4*)(src + 4 * c);
      qb[s][4 * c + 0] = (_Float16)(f.x * LOG2E);
      qb[s][4 * c + 1] = (_Float16)(f.y * LOG2E);
      qb[s][4 * c + 2] = (_Float16)(f.z * LOG2E);
      qb[s][4 * c + 3] = (_Float16)(f.w * LOG2E);
    }
  }

  // O^T accumulators: acc[ms][r] = O[q0+r15][ms*16 + r + 8*g]
  v8f acc[4];
#pragma unroll
  for (int ms = 0; ms < 4; ++ms) acc[ms] = (v8f){};
  float m_run = -INFINITY;  // running max (log2 units)
  float l_run = 0.0f;       // running sum (linear)

  const float* Kbase = k + ((size_t)bb * Nn) * Dd;
  const float* Vbase = v + ((size_t)bb * Nn) * Dd;

  const int NT = Nn / KT;  // 128 key tiles
  for (int kt = 0; kt < NT; ++kt) {
    const int kb  = kt * KT;
    const int buf = kt & 1;

    // ---- cooperative stage: K tile (f16 row-major), V tile (f16 transposed) ----
    const float4* K4 = (const float4*)(Kbase + (size_t)kb * Dd);
    const float4* V4 = (const float4*)(Vbase + (size_t)kb * Dd);
#pragma unroll
    for (int j = 0; j < 2; ++j) {
      int f   = tid + NTHR * j;         // 0..511 float4 chunks
      int row = f >> 4;                 // key row 0..31
      int c4  = f & 15;                 // float4 within row
      float4 kk = K4[row * 16 + c4];
      v4h kh = { (_Float16)kk.x, (_Float16)kk.y, (_Float16)kk.z, (_Float16)kk.w };
      *(v4h*)&Kh[buf][row][c4 * 4] = kh;           // ds_store_b64
      float4 vv = V4[row * 16 + c4];
      Vt[buf][c4 * 4 + 0][row] = (_Float16)vv.x;   // transpose scatter
      Vt[buf][c4 * 4 + 1][row] = (_Float16)vv.y;
      Vt[buf][c4 * 4 + 2][row] = (_Float16)vv.z;
      Vt[buf][c4 * 4 + 3][row] = (_Float16)vv.w;
    }
    __syncthreads();

    // prefetch next tile (global_prefetch_b8); 2x128 threads x 64B covers both 8KB tiles
    if (kt + 1 < NT) {
      const char* p = (tid < 128)
          ? (const char*)(Kbase + (size_t)(kb + KT) * Dd) + tid * 64
          : (const char*)(Vbase + (size_t)(kb + KT) * Dd) + (tid - 128) * 64;
      __builtin_prefetch(p, 0, 1);
    }

    // ---- S^T = K_tile(32x64) * (log2e*Q)^T(64x16) ----
    // Issue ALL fragment loads first so WMMAs overlap with outstanding ds_loads.
    v16h a00 = load_a16(&Kh[buf][r15][0],       g);  // keys 0..15, d 0..31
    v16h a01 = load_a16(&Kh[buf][r15][32],      g);  // keys 0..15, d 32..63
    v16h a10 = load_a16(&Kh[buf][16 + r15][0],  g);  // keys 16..31, d 0..31
    v16h a11 = load_a16(&Kh[buf][16 + r15][32], g);  // keys 16..31, d 32..63
    v8f st0 = (v8f){}, st1 = (v8f){};
    st0 = WMMA_F16(a00, qb[0], st0);
    st1 = WMMA_F16(a10, qb[0], st1);
    st0 = WMMA_F16(a01, qb[1], st0);
    st1 = WMMA_F16(a11, qb[1], st1);
    // C layout: st{t}[r] = S^T[16*t + r + 8*g][query r15]   (log2 units)

    // ---- online softmax (log2 domain): per-query stats lane-local + one xor-16 ----
    float mloc = -INFINITY;
#pragma unroll
    for (int r = 0; r < 8; ++r) mloc = fmaxf(mloc, fmaxf(st0[r], st1[r]));
    float mnew  = fmaxf(m_run, fmaxf(mloc, __shfl_xor(mloc, 16)));
    float alpha = fast_exp2(m_run - mnew);

    v8f p0, p1;
    float lloc = 0.0f;
#pragma unroll
    for (int r = 0; r < 8; ++r) {
      p0[r] = fast_exp2(st0[r] - mnew);
      p1[r] = fast_exp2(st1[r] - mnew);
      lloc += p0[r] + p1[r];
    }
    l_run = l_run * alpha + (lloc + __shfl_xor(lloc, 16));
    m_run = mnew;
#pragma unroll
    for (int ms = 0; ms < 4; ++ms)
#pragma unroll
      for (int r = 0; r < 8; ++r) acc[ms][r] *= alpha;

    // ---- P^T (C layout) -> B-operand layout: single lane-16 exchange ----
    // b[i]   (keys 16g+i)   : g==0 -> local p0    ; g==1 -> partner's p1
    // b[i+8] (keys 16g+8+i) : g==0 -> partner's p0; g==1 -> local p1
    v16h pb;
#pragma unroll
    for (int r = 0; r < 8; ++r) {
      float send = g ? p0[r] : p1[r];
      float recv = __shfl_xor(send, 16);
      float lo   = g ? recv : p0[r];
      float hi   = g ? p1[r] : recv;
      pb[r]     = (_Float16)lo;
      pb[r + 8] = (_Float16)hi;
    }

    // ---- O^T += V^T(64x32) * P^T(32x16): 4 D-slices, loads batched ahead ----
    v16h va0 = load_a16(&Vt[buf][r15][0],      g);
    v16h va1 = load_a16(&Vt[buf][16 + r15][0], g);
    v16h va2 = load_a16(&Vt[buf][32 + r15][0], g);
    v16h va3 = load_a16(&Vt[buf][48 + r15][0], g);
    acc[0] = WMMA_F16(va0, pb, acc[0]);
    acc[1] = WMMA_F16(va1, pb, acc[1]);
    acc[2] = WMMA_F16(va2, pb, acc[2]);
    acc[3] = WMMA_F16(va3, pb, acc[3]);
    // double-buffered LDS: next iteration's stage targets the other buffer;
    // its __syncthreads() guarantees everyone finished reading this one.
  }

  // ---- epilogue: softmax normalization * post-softmax scale (1/sqrt(64)) ----
  const float inv = 0.125f / l_run;
  float* op = out + ((size_t)bb * Nn + q0 + r15) * Dd + 8 * g;
#pragma unroll
  for (int ms = 0; ms < 4; ++ms) {
    float4 lo, hi;
    lo.x = acc[ms][0] * inv; lo.y = acc[ms][1] * inv;
    lo.z = acc[ms][2] * inv; lo.w = acc[ms][3] * inv;
    hi.x = acc[ms][4] * inv; hi.y = acc[ms][5] * inv;
    hi.z = acc[ms][6] * inv; hi.w = acc[ms][7] * inv;
    *(float4*)(op + ms * 16)     = lo;
    *(float4*)(op + ms * 16 + 4) = hi;
  }
}

extern "C" void kernel_launch(void* const* d_in, const int* in_sizes, int n_in,
                              void* d_out, int out_size, void* d_ws, size_t ws_size,
                              hipStream_t stream) {
  (void)in_sizes; (void)n_in; (void)out_size; (void)d_ws; (void)ws_size;
  const float* q = (const float*)d_in[0];
  const float* k = (const float*)d_in[1];
  const float* v = (const float*)d_in[2];
  float* out = (float*)d_out;
  dim3 grid(Nn / QT, Bb);
  fa_sT_kernel<<<grid, NTHR, 0, stream>>>(q, k, v, out);
}
